// VisionVLM_21294447853832
// MI455X (gfx1250) — matrix operand order repaired and checked by reading, not compiled, not run on hardware
//
#include <hip/hip_runtime.h>
#include <hip/hip_bf16.h>

// ---------------------------------------------------------------------------
// Quantized-weight MLP on gfx1250: stream int8-in-int32 weights from HBM,
// convert to f16 in registers, multiply with f16 activations via
// v_wmma_f32_16x16x32_f16. HBM-bound on weight traffic (205 MB proj), so
// weights are read exactly once per GEMM, fully coalesced, no LDS staging.
// All bounds guards are hoisted out of the K loop (EXEC all-ones for WMMA).
// ---------------------------------------------------------------------------

typedef __attribute__((ext_vector_type(16))) _Float16 v16h;
typedef __attribute__((ext_vector_type(8)))  _Float16 v8h;
typedef __attribute__((ext_vector_type(8)))  float    v8f;

namespace {

constexpr int BATCH   = 256;
constexpr int KPROJ   = 224 * 224;   // 50176
constexpr int HIDDEN  = 1024;
constexpr int NLAYERS = 8;
constexpr int NACT    = 1000;
constexpr int KSPLIT  = 16;          // proj split-K slices

union AFragU { v16h v; v8h h[2]; };

// A fragment (16-bit 16x32, ISA 7.12.2): the caller pre-offsets p by
// row*ldk + 8*(lane>=16); halves = K[+0..+7] and K[+16..+23].
__device__ __forceinline__ v16h load_a_f16(const _Float16* __restrict__ p) {
  AFragU u;
  u.h[0] = *(const v8h*)(p);
  u.h[1] = *(const v8h*)(p + 16);
  return u.v;
}

__device__ __forceinline__ v16h load_a_f32(const float* __restrict__ p) {
  float4 f0 = *(const float4*)(p);
  float4 f1 = *(const float4*)(p + 4);
  float4 f2 = *(const float4*)(p + 16);
  float4 f3 = *(const float4*)(p + 20);
  v16h a;
  a[0]=(_Float16)f0.x;  a[1]=(_Float16)f0.y;  a[2]=(_Float16)f0.z;  a[3]=(_Float16)f0.w;
  a[4]=(_Float16)f1.x;  a[5]=(_Float16)f1.y;  a[6]=(_Float16)f1.z;  a[7]=(_Float16)f1.w;
  a[8]=(_Float16)f2.x;  a[9]=(_Float16)f2.y;  a[10]=(_Float16)f2.z; a[11]=(_Float16)f2.w;
  a[12]=(_Float16)f3.x; a[13]=(_Float16)f3.y; a[14]=(_Float16)f3.z; a[15]=(_Float16)f3.w;
  return a;
}

// B fragment (16-bit 32x16): caller pre-offsets p by col*ldk + 8*(lane>=16);
// halves = K[+0..+7] and K[+16..+23] -> 4 int4 loads + cvt.
__device__ __forceinline__ v16h load_b_w8(const int* __restrict__ p) {
  int4 i0 = *(const int4*)(p);
  int4 i1 = *(const int4*)(p + 4);
  int4 i2 = *(const int4*)(p + 16);
  int4 i3 = *(const int4*)(p + 20);
  v16h b;
  b[0]=(_Float16)i0.x;  b[1]=(_Float16)i0.y;  b[2]=(_Float16)i0.z;  b[3]=(_Float16)i0.w;
  b[4]=(_Float16)i1.x;  b[5]=(_Float16)i1.y;  b[6]=(_Float16)i1.z;  b[7]=(_Float16)i1.w;
  b[8]=(_Float16)i2.x;  b[9]=(_Float16)i2.y;  b[10]=(_Float16)i2.z; b[11]=(_Float16)i2.w;
  b[12]=(_Float16)i3.x; b[13]=(_Float16)i3.y; b[14]=(_Float16)i3.z; b[15]=(_Float16)i3.w;
  return b;
}

// One generic WMMA GEMM: y[256,N] = act(x[256,K] @ W[N,K]^T * scale)
// Wave tile 16Mx32N (2 WMMAs/K-step). WG = 8 waves -> 64Mx64N block tile.
// SPLITK: each blockIdx.z computes a K-slice and stores raw f32 partials.
// Out-of-range head columns are CLAMPED for the weight stream (results for
// those lanes are discarded by the store guard / zero scale), keeping the
// K loop branch-free and EXEC all-ones for the WMMAs.
template<bool SRC_F16, bool RELU, bool OUT_F16, bool SPLITK>
__global__ __launch_bounds__(256) void wmma_gemm_kernel(
    const void* __restrict__ src, const int* __restrict__ w,
    const float* __restrict__ scale, void* __restrict__ out,
    int K, int Kc, int Nw, int Nout) {
  const int lane = threadIdx.x & 31;
  const int wave = threadIdx.x >> 5;
  const int m0 = blockIdx.y * 64 + (wave & 3) * 16;
  const int n0 = blockIdx.x * 64 + (wave >> 2) * 32;
  const int kstart = blockIdx.z * Kc;

  const int lid = lane & 15;
  const int hi  = lane >> 4;

  // Hoisted per-lane base pointers (strength-reduced inside the loop).
  const int colW0 = min(n0 + lid,      Nw - 1);
  const int colW1 = min(n0 + 16 + lid, Nw - 1);
  const int* pb0 = w + (size_t)colW0 * K + (hi << 3);
  const int* pb1 = w + (size_t)colW1 * K + (hi << 3);
  const size_t aoff = (size_t)(m0 + lid) * K + (hi << 3);
  const _Float16* pa16 = (const _Float16*)src + aoff;
  const float*    pa32 = (const float*)src + aoff;

  v8f c0 = {};
  v8f c1 = {};

#pragma unroll 2
  for (int k = kstart; k < kstart + Kc; k += 32) {
    v16h a;
    if (SRC_F16) a = load_a_f16(pa16 + k);
    else         a = load_a_f32(pa32 + k);
    v16h b0 = load_b_w8(pb0 + k);
    v16h b1 = load_b_w8(pb1 + k);
    c0 = __builtin_amdgcn_wmma_f32_16x16x32_f16(false, a, false, b0,
                                                (short)0, c0, false, false);
    c1 = __builtin_amdgcn_wmma_f32_16x16x32_f16(false, a, false, b1,
                                                (short)0, c1, false, false);
  }

  // C/D layout: VGPR r -> row m0 + r + 8*(lane>=16), col n0 + (lane&15).
  const int rbase = hi << 3;
  const int colA  = n0 + lid;
  const int colB  = colA + 16;

  if (SPLITK) {
    float* part = (float*)out + (size_t)blockIdx.z * (BATCH * HIDDEN);
#pragma unroll
    for (int r = 0; r < 8; ++r) {
      int row = m0 + rbase + r;
      part[(size_t)row * HIDDEN + colA] = c0[r];
      part[(size_t)row * HIDDEN + colB] = c1[r];
    }
  } else {
    float s0 = (colA < Nout) ? scale[colA] : 0.0f;
    float s1 = (colB < Nout) ? scale[colB] : 0.0f;
#pragma unroll
    for (int r = 0; r < 8; ++r) {
      int row = m0 + rbase + r;
      float v0 = c0[r] * s0;
      float v1 = c1[r] * s1;
      if (RELU) { v0 = fmaxf(v0, 0.0f); v1 = fmaxf(v1, 0.0f); }
      if (OUT_F16) {
        _Float16* o = (_Float16*)out;
        if (colA < Nout) o[(size_t)row * Nout + colA] = (_Float16)v0;
        if (colB < Nout) o[(size_t)row * Nout + colB] = (_Float16)v1;
      } else {
        float* o = (float*)out;
        if (colA < Nout) o[(size_t)row * Nout + colA] = v0;
        if (colB < Nout) o[(size_t)row * Nout + colB] = v1;
      }
    }
  }
}

// Reduce the KSPLIT proj partials, apply per-channel scale + ReLU, emit f16.
__global__ __launch_bounds__(256) void reduce_scale_relu_kernel(
    const float* __restrict__ part, const float* __restrict__ s,
    _Float16* __restrict__ act) {
  int i = blockIdx.x * blockDim.x + threadIdx.x;   // over 256*1024
  float v = 0.0f;
#pragma unroll
  for (int z = 0; z < KSPLIT; ++z) v += part[(size_t)z * (BATCH * HIDDEN) + i];
  v *= s[i & (HIDDEN - 1)];
  act[i] = (_Float16)fmaxf(v, 0.0f);
}

} // anonymous namespace

extern "C" void kernel_launch(void* const* d_in, const int* in_sizes, int n_in,
                              void* d_out, int out_size, void* d_ws, size_t ws_size,
                              hipStream_t stream) {
  (void)in_sizes; (void)n_in; (void)out_size; (void)ws_size;

  const float* image    = (const float*)d_in[0];
  const int*   w_proj   = (const int*)d_in[1];
  const float* s_proj   = (const float*)d_in[2];
  const int*   w_layers = (const int*)d_in[3];
  const float* s_layers = (const float*)d_in[4];
  const int*   w_head   = (const int*)d_in[5];
  const float* s_head   = (const float*)d_in[6];
  float*       logits   = (float*)d_out;

  char* ws = (char*)d_ws;
  float*    part = (float*)ws;                                    // 16 MB
  _Float16* act0 = (_Float16*)(ws + (size_t)KSPLIT * BATCH * HIDDEN * sizeof(float));
  _Float16* act1 = act0 + (size_t)BATCH * HIDDEN;

  const dim3 blk(256);

  // Proj GEMM, split-K over 16 slices: grid 16x4x16 = 1024 WGs, 8192 waves.
  wmma_gemm_kernel<false, false, false, true>
      <<<dim3(HIDDEN / 64, BATCH / 64, KSPLIT), blk, 0, stream>>>(
          image, w_proj, nullptr, part, KPROJ, KPROJ / KSPLIT, HIDDEN, HIDDEN);

  // Partial-sum reduce + scale + ReLU -> f16 activations.
  reduce_scale_relu_kernel<<<dim3((BATCH * HIDDEN) / 256), blk, 0, stream>>>(
      part, s_proj, act0);

  // 8 fused hidden layers (scale + ReLU + f16 store in epilogue).
  _Float16* cur = act0;
  _Float16* nxt = act1;
  for (int l = 0; l < NLAYERS; ++l) {
    wmma_gemm_kernel<true, true, true, false>
        <<<dim3(HIDDEN / 64, BATCH / 64, 1), blk, 0, stream>>>(
            cur, w_layers + (size_t)l * HIDDEN * HIDDEN,
            s_layers + (size_t)l * HIDDEN, nxt,
            HIDDEN, HIDDEN, HIDDEN, HIDDEN);
    _Float16* t = cur; cur = nxt; nxt = t;
  }

  // Head: f32 logits, no ReLU, N=1000 edge guards (16 N-blocks cover 1024).
  wmma_gemm_kernel<true, false, false, false>
      <<<dim3(16, BATCH / 64, 1), blk, 0, stream>>>(
          cur, w_head, s_head, logits, HIDDEN, HIDDEN, NACT, NACT);
}
